// VQVAE_28707561406861
// MI455X (gfx1250) — matrix operand order repaired and checked by reading, not compiled, not run on hardware
//
#include <hip/hip_runtime.h>
#include <hip/hip_fp16.h>

typedef _Float16 f16;
typedef __attribute__((ext_vector_type(16))) _Float16 v16h;
typedef __attribute__((ext_vector_type(8)))  float    v8f;

struct Taps { int n; int dy[9]; int dx[9]; };

// ---------------------------------------------------------------------------
// B fragment: Wt is [Npad][K32] f16, row-major per output channel.
// B 32x16 layout (matches A): elem e -> K = k0 + (e&7) + 8*(lane>=16) + 16*(e>=8);
// column N = lane & 15.
// => each lane reads two 8-half runs = two b128 loads.
// ---------------------------------------------------------------------------
__device__ inline v16h load_B(const f16* Wt, int K32, int nrow, int k0, int lane) {
    const f16* p = Wt + (size_t)nrow * K32 + k0 + ((lane >> 4) << 3);
    v16h b;
    ((float4*)&b)[0] = ((const float4*)p)[0];
    ((float4*)&b)[1] = ((const float4*)p)[2];
    return b;
}

// ---------------------------------------------------------------------------
// A fragment (vector path, Cin % 8 == 0).
// ISA 16-bit A 16x32 layout: lane L (L<16) row M=L holds K {0..7,16..23},
// lane L+16 holds K {8..15,24..31}; elem e -> K = (e&7) + 8*(L>=16) + 16*(e>=8).
// Each 8-run shares one tap -> one b128 load with zero-fill on OOB/pad.
// ---------------------------------------------------------------------------
__device__ inline v16h load_A_vec(const f16* X, int bb, int yh, int xh,
                                  int Hin, int Win, int Cin, int instep,
                                  int k0, int Kreal, const Taps& tp, int lane) {
    v16h a; f16* ap = (f16*)&a;
#pragma unroll
    for (int h = 0; h < 2; ++h) {
        int kr = k0 + ((lane >> 4) << 3) + (h << 4);
        bool got = false;
        if (kr < Kreal) {
            int tap = kr / Cin;
            int ci  = kr - tap * Cin;
            int ih  = yh * instep + tp.dy[tap];
            int iw  = xh * instep + tp.dx[tap];
            if (ih >= 0 && ih < Hin && iw >= 0 && iw < Win) {
                const float4* p = (const float4*)(X + (((size_t)bb * Hin + ih) * Win + iw) * Cin + ci);
                ((float4*)ap)[h] = *p;
                got = true;
            }
        }
        if (!got) {
#pragma unroll
            for (int e = 0; e < 8; ++e) ap[h * 8 + e] = (f16)0;
        }
    }
    return a;
}

// A fragment, element-wise f32 source (conv1: Cin=3, K=27)
__device__ inline v16h load_A_elem(const float* X, int bb, int yh, int xh,
                                   int Hin, int Win, int Cin, int instep,
                                   int k0, int Kreal, const Taps& tp, int lane) {
    v16h a; f16* ap = (f16*)&a;
#pragma unroll
    for (int e = 0; e < 16; ++e) {
        int k = k0 + (e & 7) + ((lane >> 4) << 3) + ((e >> 3) << 4);
        float v = 0.f;
        if (k < Kreal) {
            int tap = k / Cin, ci = k - tap * Cin;
            int ih = yh * instep + tp.dy[tap];
            int iw = xh * instep + tp.dx[tap];
            if (ih >= 0 && ih < Hin && iw >= 0 && iw < Win)
                v = X[(((size_t)bb * Hin + ih) * Win + iw) * Cin + ci];
        }
        ap[e] = (f16)v;
    }
    return a;
}

// ---------------------------------------------------------------------------
// Generic implicit-im2col GEMM conv.  One wave = 16 output pixels x Npad chans.
// Class output grid (Hc,Wc); write position oh = yh*wmul+wy, ow = xh*wmul+wx.
// NT = Npad/16 accumulator tiles (compile-time -> registers).
// ---------------------------------------------------------------------------
template<int NT, bool ELEM, bool OUTF32, bool RELU>
__global__ __launch_bounds__(256) void conv_gemm_kernel(
    const void* __restrict__ Xv, int Hin, int Win, int Cin,
    const f16* __restrict__ Wt, const float* __restrict__ bias,
    void* __restrict__ Yv,
    int Hout, int Wout, int Cout, int K32, int Kreal,
    int Hc, int Wc, int instep, int wmul, int wy, int wx, Taps tp)
{
    int lane = threadIdx.x & 31;
    int wave = threadIdx.x >> 5;
    long tile = (long)blockIdx.x * 8 + wave;
    long p0 = tile << 4;
    // Wc is a multiple of 16 -> all 16 pixels of a tile share (b, yh)
    int xh0 = (int)(p0 % Wc);
    long t  = p0 / Wc;
    int yh  = (int)(t % Hc);
    int bb  = (int)(t / Hc);
    int xh  = xh0 + (lane & 15);

    v8f acc[NT];
#pragma unroll
    for (int q = 0; q < NT; ++q)
#pragma unroll
        for (int r = 0; r < 8; ++r) acc[q][r] = 0.f;

    for (int k0 = 0; k0 < K32; k0 += 32) {
        v16h a;
        if constexpr (ELEM)
            a = load_A_elem((const float*)Xv, bb, yh, xh, Hin, Win, Cin, instep, k0, Kreal, tp, lane);
        else
            a = load_A_vec((const f16*)Xv, bb, yh, xh, Hin, Win, Cin, instep, k0, Kreal, tp, lane);
#pragma unroll
        for (int q = 0; q < NT; ++q) {
            v16h bf = load_B(Wt, K32, (q << 4) + (lane & 15), k0, lane);
            acc[q] = __builtin_amdgcn_wmma_f32_16x16x32_f16(
                false, a, false, bf, (short)0, acc[q], false, false);
        }
    }

    // C/D layout: elem r of v8f -> row m = r + 8*(lane>=16), col n = lane&15
    int half = lane >> 4;
    int oh = yh * wmul + wy;
#pragma unroll
    for (int q = 0; q < NT; ++q) {
        int co = (q << 4) + (lane & 15);
        if (co >= Cout) continue;
        float bv = bias ? bias[co] : 0.f;
#pragma unroll
        for (int r = 0; r < 8; ++r) {
            int mm  = r + (half << 3);
            int ow  = (xh0 + mm) * wmul + wx;
            float v = acc[q][r] + bv;
            if (RELU) v = fmaxf(v, 0.f);
            size_t oidx = (((size_t)bb * Hout + oh) * Wout + ow) * (size_t)Cout + co;
            if constexpr (OUTF32) ((float*)Yv)[oidx] = v;
            else                  ((f16*)Yv)[oidx]   = (f16)v;
        }
    }
}

// ---------------------------------------------------------------------------
// VQ: scores S = F @ CB via WMMA (K=16 zero-padded to 32); argmin over 512
// codes of (||c||^2 - 2 f.c); gather winning codebook row into q (f16).
// ---------------------------------------------------------------------------
__global__ __launch_bounds__(256) void vq_kernel(
    const f16* __restrict__ Z, const f16* __restrict__ WtQ,
    const f16* __restrict__ cbT, const float* __restrict__ cnorm,
    f16* __restrict__ Q)
{
    int lane = threadIdx.x & 31;
    int wave = threadIdx.x >> 5;
    long tile = (long)blockIdx.x * 8 + wave;
    long p0 = tile << 4;
    long p  = p0 + (lane & 15);

    v16h a; f16* ap = (f16*)&a;
    {
        int kr = (lane >> 4) << 3;                     // 0 or 8 (K dim of z)
        ((float4*)ap)[0] = *(const float4*)(Z + p * 16 + kr);
#pragma unroll
        for (int e = 8; e < 16; ++e) ap[e] = (f16)0;   // K 16..31 padded
    }

    float bs[8]; int bi[8];
#pragma unroll
    for (int r = 0; r < 8; ++r) { bs[r] = 3.4e38f; bi[r] = 0; }

    for (int t = 0; t < 32; ++t) {
        v16h bf = load_B(WtQ, 32, (t << 4) + (lane & 15), 0, lane);
        v8f c;
#pragma unroll
        for (int r = 0; r < 8; ++r) c[r] = 0.f;
        c = __builtin_amdgcn_wmma_f32_16x16x32_f16(false, a, false, bf, (short)0, c, false, false);
        int n = (t << 4) + (lane & 15);
        float cn = cnorm[n];
#pragma unroll
        for (int r = 0; r < 8; ++r) {
            float s = cn - 2.f * c[r];
            if (s < bs[r] || (s == bs[r] && n < bi[r])) { bs[r] = s; bi[r] = n; }
        }
    }
    // reduce across the 16 lanes sharing each row-half (xor 1,2,4,8 stays in-half)
#pragma unroll
    for (int step = 1; step < 16; step <<= 1) {
#pragma unroll
        for (int r = 0; r < 8; ++r) {
            float os = __shfl_xor(bs[r], step, 32);
            int   oi = __shfl_xor(bi[r], step, 32);
            if (os < bs[r] || (os == bs[r] && oi < bi[r])) { bs[r] = os; bi[r] = oi; }
        }
    }
    int c = lane & 15;
    if (c < 8) {
        int mm = c + ((lane >> 4) << 3);
        long pp = p0 + mm;
        const float4* src = (const float4*)(cbT + (size_t)bi[c] * 16);
        float4* dst = (float4*)(Q + pp * 16);
        dst[0] = src[0];
        dst[1] = src[1];
    }
}

// ---------------------------------------------------------------------------
// Weight packing: HWIO f32 -> [Npad][K32] f16, tap order given by tp (dy=kh,dx=kw)
// ---------------------------------------------------------------------------
__global__ void pack_w_kernel(const float* __restrict__ src, f16* __restrict__ dst,
                              int KW, int Cin, int Cout, int Npad, int K32, int Kreal, Taps tp)
{
    int idx = blockIdx.x * blockDim.x + threadIdx.x;
    if (idx >= Npad * K32) return;
    int n = idx / K32, k = idx - n * K32;
    float v = 0.f;
    if (n < Cout && k < Kreal) {
        int tap = k / Cin, ci = k - tap * Cin;
        int kh = tp.dy[tap], kw = tp.dx[tap];
        v = src[(((size_t)kh * KW + kw) * Cin + ci) * Cout + n];
    }
    dst[idx] = (f16)v;
}

__global__ void pack_cb_kernel(const float* __restrict__ cb, f16* __restrict__ WtQ,
                               f16* __restrict__ cbT, float* __restrict__ cnorm)
{
    int n = blockIdx.x * blockDim.x + threadIdx.x;
    if (n >= 512) return;
    float s = 0.f;
    for (int k = 0; k < 16; ++k) {
        float v = cb[k * 512 + n];
        s += v * v;
        WtQ[n * 32 + k] = (f16)v;
        cbT[n * 16 + k] = (f16)v;
    }
    for (int k = 16; k < 32; ++k) WtQ[n * 32 + k] = (f16)0;
    cnorm[n] = s;
}

// ---------------------------------------------------------------------------
// Host side
// ---------------------------------------------------------------------------
static Taps full3(int off) {
    Taps t; t.n = 9; int i = 0;
    for (int kh = 0; kh < 3; ++kh)
        for (int kw = 0; kw < 3; ++kw, ++i) { t.dy[i] = kh + off; t.dx[i] = kw + off; }
    return t;
}
static Taps tap1() { Taps t; t.n = 1; t.dy[0] = 0; t.dx[0] = 0; return t; }

// convT stride-2 parity class (jax conv_transpose SAME k=3 s=2: pad(2,1), no flip):
// even parity -> taps (off=-1,kh=0),(off=0,kh=2); odd -> (off=0,kh=1)
static void build_class(int py, int px, Taps& pk, Taps& cv, int& Kreal, int Cin) {
    int offy[2], khy[2], ny, offx[2], khx[2], nx;
    if (py) { ny = 1; offy[0] = 0;  khy[0] = 1; }
    else    { ny = 2; offy[0] = -1; khy[0] = 0; offy[1] = 0; khy[1] = 2; }
    if (px) { nx = 1; offx[0] = 0;  khx[0] = 1; }
    else    { nx = 2; offx[0] = -1; khx[0] = 0; offx[1] = 0; khx[1] = 2; }
    int i = 0;
    for (int a = 0; a < ny; ++a)
        for (int b = 0; b < nx; ++b, ++i) {
            pk.dy[i] = khy[a]; pk.dx[i] = khx[b];
            cv.dy[i] = offy[a]; cv.dx[i] = offx[b];
        }
    pk.n = cv.n = i;
    Kreal = i * Cin;
}

static inline size_t alignUp(size_t x, size_t a) { return (x + a - 1) & ~(a - 1); }
static inline int roundK32(int k) { return (k + 31) & ~31; }

extern "C" void kernel_launch(void* const* d_in, const int* in_sizes, int n_in,
                              void* d_out, int out_size, void* d_ws, size_t ws_size,
                              hipStream_t stream) {
    const float* x   = (const float*)d_in[0];
    const float* w1  = (const float*)d_in[1];  const float* b1  = (const float*)d_in[2];
    const float* w2  = (const float*)d_in[3];  const float* b2  = (const float*)d_in[4];
    const float* w3  = (const float*)d_in[5];  const float* b3  = (const float*)d_in[6];
    const float* cb  = (const float*)d_in[7];
    const float* t1  = (const float*)d_in[8];  const float* tb1 = (const float*)d_in[9];
    const float* t2  = (const float*)d_in[10]; const float* tb2 = (const float*)d_in[11];
    const float* t3  = (const float*)d_in[12]; const float* tb3 = (const float*)d_in[13];
    float* out = (float*)d_out;

    char* ws = (char*)d_ws;
    size_t off = 0;
    auto bump = [&](size_t bytes) { size_t o = off; off = alignUp(off + bytes, 256); return o; };

    // big activation buffers (reused)
    size_t offP = bump((size_t)32 * 256 * 256 * 32 * 2);  // 134 MB: h1 then d2
    size_t offQ = bump((size_t)32 * 128 * 128 * 64 * 2);  //  67 MB: h2 then d1
    size_t offZ = bump((size_t)131072 * 16 * 2);          // z (f16)
    size_t offG = bump((size_t)131072 * 16 * 2);          // q (f16)

    // packed weights
    size_t oW1  = bump(32 * 32 * 2);
    size_t oW2  = bump((size_t)64 * 288 * 2);
    size_t oW3  = bump(16 * 64 * 2);
    size_t oWQ  = bump(512 * 32 * 2);
    size_t oCBT = bump(512 * 16 * 2);
    size_t oCN  = bump(512 * 4);

    Taps pkT1[4], cvT1[4], pkT2[4], cvT2[4];
    int  kT1[4], kT2[4], k32T1[4], k32T2[4];
    size_t oT1[4], oT2[4];
    for (int c = 0; c < 4; ++c) {
        int py = c >> 1, px = c & 1;
        build_class(py, px, pkT1[c], cvT1[c], kT1[c], 16);
        build_class(py, px, pkT2[c], cvT2[c], kT2[c], 64);
        k32T1[c] = roundK32(kT1[c]);
        k32T2[c] = roundK32(kT2[c]);
        oT1[c] = bump((size_t)64 * k32T1[c] * 2);
        oT2[c] = bump((size_t)32 * k32T2[c] * 2);
    }
    size_t oT3 = bump((size_t)16 * 288 * 2);
    (void)ws_size; (void)in_sizes; (void)n_in; (void)out_size;

    f16* P   = (f16*)(ws + offP);  f16* Q   = (f16*)(ws + offQ);
    f16* Zb  = (f16*)(ws + offZ);  f16* Gq  = (f16*)(ws + offG);
    f16* Wt1 = (f16*)(ws + oW1);   f16* Wt2 = (f16*)(ws + oW2);   f16* Wt3 = (f16*)(ws + oW3);
    f16* WtQ = (f16*)(ws + oWQ);   f16* CBT = (f16*)(ws + oCBT);  float* CN = (float*)(ws + oCN);

    Taps f3  = full3(0);     // pack taps & encoder conv taps (pad_lo = 0 for s2 SAME)
    Taps f3m = full3(-1);    // convT3 conv taps (pad 1)
    Taps one = tap1();

    auto packGrid = [](int n) { return dim3((n + 255) / 256); };

    // ---- pack weights + codebook --------------------------------------
    pack_cb_kernel<<<dim3(2), 256, 0, stream>>>(cb, WtQ, CBT, CN);
    pack_w_kernel<<<packGrid(32 * 32),   256, 0, stream>>>(w1, Wt1, 3, 3,  32, 32, 32, 27,  f3);
    pack_w_kernel<<<packGrid(64 * 288),  256, 0, stream>>>(w2, Wt2, 3, 32, 64, 64, 288, 288, f3);
    pack_w_kernel<<<packGrid(16 * 64),   256, 0, stream>>>(w3, Wt3, 1, 64, 16, 16, 64, 64,  one);
    for (int c = 0; c < 4; ++c) {
        pack_w_kernel<<<packGrid(64 * k32T1[c]), 256, 0, stream>>>(
            t1, (f16*)(ws + oT1[c]), 3, 16, 64, 64, k32T1[c], kT1[c], pkT1[c]);
        pack_w_kernel<<<packGrid(32 * k32T2[c]), 256, 0, stream>>>(
            t2, (f16*)(ws + oT2[c]), 3, 64, 32, 32, k32T2[c], kT2[c], pkT2[c]);
    }
    pack_w_kernel<<<packGrid(16 * 288), 256, 0, stream>>>(t3, (f16*)(ws + oT3), 3, 32, 3, 16, 288, 288, f3);

    // ---- encoder ------------------------------------------------------
    // conv1: [32,256,256,3] -> h1 [32,128,128,32], s2, ReLU  (M=524288)
    conv_gemm_kernel<2, true, false, true><<<dim3(4096), 256, 0, stream>>>(
        x, 256, 256, 3, Wt1, b1, P, 128, 128, 32, 32, 27,
        128, 128, /*instep*/2, /*wmul*/1, 0, 0, f3);
    // conv2: h1 -> h2 [32,64,64,64], s2, ReLU  (M=131072)
    conv_gemm_kernel<4, false, false, true><<<dim3(1024), 256, 0, stream>>>(
        P, 128, 128, 32, Wt2, b2, Q, 64, 64, 64, 288, 288,
        64, 64, 2, 1, 0, 0, f3);
    // conv3 (1x1): h2 -> z [32,64,64,16], no ReLU
    conv_gemm_kernel<1, false, false, false><<<dim3(1024), 256, 0, stream>>>(
        Q, 64, 64, 64, Wt3, b3, Zb, 64, 64, 16, 64, 64,
        64, 64, 1, 1, 0, 0, one);

    // ---- vector quantisation (forward == gather of nearest code) -----
    vq_kernel<<<dim3(1024), 256, 0, stream>>>(Zb, WtQ, CBT, CN, Gq);

    // ---- decoder ------------------------------------------------------
    // convT1: q -> d1 [32,128,128,64], ReLU; 4 parity classes (M=131072 each)
    for (int c = 0; c < 4; ++c) {
        int py = c >> 1, px = c & 1;
        conv_gemm_kernel<4, false, false, true><<<dim3(1024), 256, 0, stream>>>(
            Gq, 64, 64, 16, (f16*)(ws + oT1[c]), tb1, Q, 128, 128, 64,
            k32T1[c], kT1[c], 64, 64, 1, /*wmul*/2, py, px, cvT1[c]);
    }
    // convT2: d1 -> d2 [32,256,256,32], ReLU (M=524288 each)
    for (int c = 0; c < 4; ++c) {
        int py = c >> 1, px = c & 1;
        conv_gemm_kernel<2, false, false, true><<<dim3(4096), 256, 0, stream>>>(
            Q, 128, 128, 64, (f16*)(ws + oT2[c]), tb2, P, 256, 256, 32,
            k32T2[c], kT2[c], 128, 128, 1, 2, py, px, cvT2[c]);
    }
    // convT3 (s1 == plain SAME conv, no flip): d2 -> out [32,256,256,3] f32
    conv_gemm_kernel<1, false, true, false><<<dim3(16384), 256, 0, stream>>>(
        P, 256, 256, 32, (f16*)(ws + oT3), tb3, out, 256, 256, 3, 288, 288,
        256, 256, 1, 1, 0, 0, f3m);
}
